// DecoderBlock_13443247636967
// MI455X (gfx1250) — compile-verified, matrix operand order checked by reading, not run on hardware
//
#include <hip/hip_runtime.h>

// ---------------------------------------------------------------------------
// Types for CDNA5 WMMA (wave32): v_wmma_f32_16x16x32_bf16
// ---------------------------------------------------------------------------
typedef __attribute__((ext_vector_type(16))) __bf16 v16bf;
typedef __attribute__((ext_vector_type(8)))  __bf16 v8bf;
typedef __attribute__((ext_vector_type(8)))  float  v8f;
typedef __attribute__((ext_vector_type(4)))  float  v4f;

#define DEV __device__ __forceinline__

// Problem constants
static constexpr int   kB   = 4;
static constexpr int   kT   = 2048;
static constexpr int   kD   = 1024;
static constexpr int   kH   = 16;
static constexpr int   kHS  = 64;
static constexpr int   kM   = kB * kT;            // 8192 rows
static constexpr float kSCALE = 1.0f / 1073741824.0f;  // 64^-5 = 2^-30 (faithful to source)
static constexpr float kEPS   = 1e-5f;

// float -> bf16, round-to-nearest-even
DEV __bf16 f2bf(float x) {
  unsigned u = __builtin_bit_cast(unsigned, x);
  u = (u + 0x7FFFu + ((u >> 16) & 1u)) >> 16;
  unsigned short s = (unsigned short)u;
  return __builtin_bit_cast(__bf16, s);
}

// Load a 16x32 bf16 tile (row-major, `stride` elements between rows) into the
// A/B fragment layout for V_WMMA_*_16X16X32_BF16:
//   lanes 0-15 : row = lane,    K elements {0..7, 16..23}
//   lanes 16-31: row = lane-16, K elements {8..15, 24..31}
// Both halves are two contiguous 16-byte loads.
DEV v16bf load_frag(const __bf16* base, int stride, int lane) {
  const __bf16* p = base + (size_t)(lane & 15) * stride + ((lane >> 4) << 3);
  v8bf lo = *reinterpret_cast<const v8bf*>(p);
  v8bf hi = *reinterpret_cast<const v8bf*>(p + 16);
  v16bf f;
#pragma unroll
  for (int i = 0; i < 8; ++i) { f[i] = lo[i]; f[i + 8] = hi[i]; }
  return f;
}

DEV v8f wmma_bf16(v16bf a, v16bf b, v8f c) {
  return __builtin_amdgcn_wmma_f32_16x16x32_bf16(
      /*neg_a=*/false, a, /*neg_b=*/false, b,
      /*c_mod=*/(short)0, c, /*reuse_a=*/false, /*reuse_b=*/false);
}

// ---------------------------------------------------------------------------
// Elementwise f32 -> bf16 convert
// ---------------------------------------------------------------------------
__global__ __launch_bounds__(256) void convert_bf16_k(
    const float* __restrict__ in, __bf16* __restrict__ out, int n) {
  int i = blockIdx.x * 256 + threadIdx.x;
  if (i < n) out[i] = f2bf(in[i]);
}

// Transpose [rows,cols] f32 -> [cols,rows] bf16 (batched over grid.y)
__global__ __launch_bounds__(256) void transpose_bf16_k(
    const float* __restrict__ in, __bf16* __restrict__ out, int rows, int cols) {
  size_t base = (size_t)blockIdx.y * rows * cols;
  int idx = blockIdx.x * 256 + threadIdx.x;
  if (idx < rows * cols) {
    int k = idx / cols, n = idx % cols;
    out[base + (size_t)n * rows + k] = f2bf(in[base + idx]);
  }
}

// ---------------------------------------------------------------------------
// K/V projection: K[bh,t,s] = X @ Wk[h];  VT[bh,s,t] = (X @ Wv[h])^T
// Block = 128 threads (4 waves); each wave computes 32 rows x 64 cols for
// both K and V (16 WMMA accumulators; 16 WMMAs per 10 fragment loads).
// grid = (M/128, H).
// ---------------------------------------------------------------------------
__global__ __launch_bounds__(128) void proj_kv_k(
    const __bf16* __restrict__ X, const __bf16* __restrict__ WkT,
    const __bf16* __restrict__ WvT, __bf16* __restrict__ K,
    __bf16* __restrict__ VT) {
  const int lane = threadIdx.x & 31, wave = threadIdx.x >> 5;
  const int h  = blockIdx.y;
  const int m0 = blockIdx.x * 128 + wave * 32;     // global row in [0, 8192)
  const __bf16* wk = WkT + (size_t)h * kHS * kD;   // [64 x 1024] row-major (N,K)
  const __bf16* wv = WvT + (size_t)h * kHS * kD;

  v8f accK[2][4] = {}, accV[2][4] = {};
  for (int k = 0; k < kD; k += 32) {
    v16bf a0 = load_frag(X + (size_t)m0 * kD + k,        kD, lane);
    v16bf a1 = load_frag(X + (size_t)(m0 + 16) * kD + k, kD, lane);
#pragma unroll
    for (int f = 0; f < 4; ++f) {
      v16bf bk = load_frag(wk + (size_t)(f * 16) * kD + k, kD, lane);
      accK[0][f] = wmma_bf16(a0, bk, accK[0][f]);
      accK[1][f] = wmma_bf16(a1, bk, accK[1][f]);
      v16bf bv = load_frag(wv + (size_t)(f * 16) * kD + k, kD, lane);
      accV[0][f] = wmma_bf16(a0, bv, accV[0][f]);
      accV[1][f] = wmma_bf16(a1, bv, accV[1][f]);
    }
  }

  const int half = lane >> 4, col = lane & 15;
  const int bidx = m0 / kT;                        // 128-row tiles never cross batch
  const int bh   = bidx * kH + h;
  const int tl0  = m0 % kT;
#pragma unroll
  for (int mi = 0; mi < 2; ++mi)
#pragma unroll
    for (int f = 0; f < 4; ++f)
#pragma unroll
      for (int r = 0; r < 8; ++r) {
        int t = tl0 + mi * 16 + r + 8 * half;
        int s = f * 16 + col;
        K [((size_t)bh * kT + t) * kHS + s] = f2bf(accK[mi][f][r]);
        VT[((size_t)bh * kHS + s) * kT + t] = f2bf(accV[mi][f][r]);
      }
}

// ---------------------------------------------------------------------------
// Causal flash attention with q == k (source bug).
// Block = 256 threads (8 waves); each wave owns 16 query rows; key chunks of
// 32; online softmax; P restaged via per-wave LDS slab (no inter-wave sync).
// grid = (T/128, B*H).
// ---------------------------------------------------------------------------
__global__ __launch_bounds__(256) void attention_k(
    const __bf16* __restrict__ K, const __bf16* __restrict__ VT,
    __bf16* __restrict__ O) {
  alignas(16) __shared__ __bf16 Plds[8][16][32];

  const int lane = threadIdx.x & 31, wave = threadIdx.x >> 5;
  const int bh  = blockIdx.y;
  const int qm0 = blockIdx.x * 128 + wave * 16;
  const int half = lane >> 4, col = lane & 15;

  const __bf16* Kb = K  + (size_t)bh * kT * kHS;   // [2048 x 64] row-major
  const __bf16* Vb = VT + (size_t)bh * kHS * kT;   // [64 x 2048] row-major

  // Q fragments (q == k): rows qm0..qm0+15, K-depth 0..31 and 32..63
  v16bf qa0 = load_frag(Kb + (size_t)qm0 * kHS,      kHS, lane);
  v16bf qa1 = load_frag(Kb + (size_t)qm0 * kHS + 32, kHS, lane);

  v8f acc[4] = {};
  float mrow[8], lrow[8];
#pragma unroll
  for (int r = 0; r < 8; ++r) { mrow[r] = -1e30f; lrow[r] = 0.0f; }

  const int nchunks = (qm0 + 16 + 31) >> 5;        // cover keys 0..qm0+15
  for (int c = 0; c < nchunks; ++c) {
    const int u0 = c * 32;
    // S = (Q K^T) over two 16x16 key tiles; Bt for Q.K^T is K itself.
    v16bf b00 = load_frag(Kb + (size_t)u0 * kHS,             kHS, lane);
    v16bf b01 = load_frag(Kb + (size_t)u0 * kHS + 32,        kHS, lane);
    v16bf b10 = load_frag(Kb + (size_t)(u0 + 16) * kHS,      kHS, lane);
    v16bf b11 = load_frag(Kb + (size_t)(u0 + 16) * kHS + 32, kHS, lane);
    v8f z = {};
    v8f s0 = wmma_bf16(qa1, b01, wmma_bf16(qa0, b00, z));
    v8f s1 = wmma_bf16(qa1, b11, wmma_bf16(qa0, b10, z));

#pragma unroll
    for (int r = 0; r < 8; ++r) {
      const int q = qm0 + r + 8 * half;
      float v0 = (u0 + col      <= q) ? s0[r] * kSCALE : -1e30f;
      float v1 = (u0 + 16 + col <= q) ? s1[r] * kSCALE : -1e30f;
      float cm = fmaxf(v0, v1);
      cm = fmaxf(cm, __shfl_xor(cm, 1, 32));
      cm = fmaxf(cm, __shfl_xor(cm, 2, 32));
      cm = fmaxf(cm, __shfl_xor(cm, 4, 32));
      cm = fmaxf(cm, __shfl_xor(cm, 8, 32));
      float nm    = fmaxf(mrow[r], cm);
      float alpha = __expf(mrow[r] - nm);
      mrow[r] = nm;
      float p0 = __expf(v0 - nm);
      float p1 = __expf(v1 - nm);
      float rs = p0 + p1;
      rs += __shfl_xor(rs, 1, 32);
      rs += __shfl_xor(rs, 2, 32);
      rs += __shfl_xor(rs, 4, 32);
      rs += __shfl_xor(rs, 8, 32);
      lrow[r] = lrow[r] * alpha + rs;
#pragma unroll
      for (int f = 0; f < 4; ++f) acc[f][r] *= alpha;
      const int Mr = r + 8 * half;                 // C-layout row
      Plds[wave][Mr][col]      = f2bf(p0);
      Plds[wave][Mr][col + 16] = f2bf(p1);
    }

    // Intra-wave LDS RAW: LDS ops are in-order per wave; fence the compiler
    // and drain DScnt before re-reading as an A fragment.
    __asm__ volatile("s_wait_dscnt 0x0" ::: "memory");
    v16bf pa = load_frag(&Plds[wave][0][0], 32, lane);
#pragma unroll
    for (int f = 0; f < 4; ++f) {
      v16bf vb = load_frag(Vb + (size_t)(f * 16) * kT + u0, kT, lane);
      acc[f] = wmma_bf16(pa, vb, acc[f]);
    }
  }

  const int b = bh >> 4, h = bh & 15;
#pragma unroll
  for (int r = 0; r < 8; ++r) {
    float inv = 1.0f / lrow[r];
    int t = qm0 + r + 8 * half;
    size_t rowoff = ((size_t)b * kT + t) * kD + (size_t)h * kHS;
#pragma unroll
    for (int f = 0; f < 4; ++f)
      O[rowoff + f * 16 + col] = f2bf(acc[f][r] * inv);
  }
}

// ---------------------------------------------------------------------------
// Generic bf16 WMMA GEMM: C[M,1024] = A[M,1024] @ Bt[1024,1024]^T
// with fused bias / up to two f32 residual adds / ReLU / f32 or bf16 store.
// Block = 256 threads (8 waves as 2x4); wave tile 64x64 (16 WMMAs per 8
// fragment loads per k-step); WG tile 128x256. grid = (M/128, 1024/256).
// ---------------------------------------------------------------------------
__global__ __launch_bounds__(256) void gemm_bf16_k(
    const __bf16* __restrict__ A, const __bf16* __restrict__ Bt,
    const float* __restrict__ bias, const float* __restrict__ res1,
    const float* __restrict__ res2, float* __restrict__ outF,
    __bf16* __restrict__ outBF, int relu) {
  const int lane = threadIdx.x & 31, wave = threadIdx.x >> 5;
  const int wm = wave >> 2, wn = wave & 3;
  const int m0 = blockIdx.x * 128 + wm * 64;
  const int n0 = blockIdx.y * 256 + wn * 64;

  v8f acc[4][4] = {};
  for (int k = 0; k < kD; k += 32) {
    v16bf a[4];
#pragma unroll
    for (int mi = 0; mi < 4; ++mi)
      a[mi] = load_frag(A + (size_t)(m0 + mi * 16) * kD + k, kD, lane);
#pragma unroll
    for (int f = 0; f < 4; ++f) {
      v16bf b = load_frag(Bt + (size_t)(n0 + f * 16) * kD + k, kD, lane);
#pragma unroll
      for (int mi = 0; mi < 4; ++mi)
        acc[mi][f] = wmma_bf16(a[mi], b, acc[mi][f]);
    }
  }

  const int half = lane >> 4, col = lane & 15;
#pragma unroll
  for (int mi = 0; mi < 4; ++mi)
#pragma unroll
    for (int f = 0; f < 4; ++f)
#pragma unroll
      for (int r = 0; r < 8; ++r) {
        int row = m0 + mi * 16 + r + 8 * half;
        int cn  = n0 + f * 16 + col;
        size_t idx = (size_t)row * kD + cn;
        float v = acc[mi][f][r] + bias[cn];
        if (res1) v += res1[idx];
        if (res2) v += res2[idx];
        if (relu) v = fmaxf(v, 0.0f);
        if (outF)  outF[idx]  = v;
        if (outBF) outBF[idx] = f2bf(v);
      }
}

// ---------------------------------------------------------------------------
// LayerNorm over D=1024; one row per 256-thread block; optional f32 + bf16 out.
// ---------------------------------------------------------------------------
__global__ __launch_bounds__(256) void layernorm_k(
    const float* __restrict__ in, const float* __restrict__ g,
    const float* __restrict__ b, float* __restrict__ outF,
    __bf16* __restrict__ outBF) {
  __shared__ float red[2][8];
  const int row = blockIdx.x;
  const int tid = threadIdx.x;
  const float* p = in + (size_t)row * kD;
  v4f x = *reinterpret_cast<const v4f*>(p + tid * 4);
  float s1 = x[0] + x[1] + x[2] + x[3];
  float s2 = x[0]*x[0] + x[1]*x[1] + x[2]*x[2] + x[3]*x[3];
#pragma unroll
  for (int m = 16; m; m >>= 1) {
    s1 += __shfl_xor(s1, m, 32);
    s2 += __shfl_xor(s2, m, 32);
  }
  const int wave = tid >> 5, lane = tid & 31;
  if (lane == 0) { red[0][wave] = s1; red[1][wave] = s2; }
  __syncthreads();
  if (wave == 0) {
    float a1 = (lane < 8) ? red[0][lane] : 0.0f;
    float a2 = (lane < 8) ? red[1][lane] : 0.0f;
#pragma unroll
    for (int m = 4; m; m >>= 1) {
      a1 += __shfl_xor(a1, m, 32);
      a2 += __shfl_xor(a2, m, 32);
    }
    if (lane == 0) { red[0][0] = a1; red[1][0] = a2; }
  }
  __syncthreads();
  float mean = red[0][0] * (1.0f / kD);
  float var  = red[1][0] * (1.0f / kD) - mean * mean;
  float rstd = rsqrtf(var + kEPS);
#pragma unroll
  for (int i = 0; i < 4; ++i) {
    int cn = tid * 4 + i;
    float v = (x[i] - mean) * rstd * g[cn] + b[cn];
    if (outF)  outF[(size_t)row * kD + cn]  = v;
    if (outBF) outBF[(size_t)row * kD + cn] = f2bf(v);
  }
}

// ---------------------------------------------------------------------------
// Host orchestration
// ---------------------------------------------------------------------------
extern "C" void kernel_launch(void* const* d_in, const int* in_sizes, int n_in,
                              void* d_out, int out_size, void* d_ws, size_t ws_size,
                              hipStream_t stream) {
  (void)in_sizes; (void)n_in; (void)out_size; (void)ws_size;
  const float* x   = (const float*)d_in[0];
  const float* Wk  = (const float*)d_in[1];
  const float* Wv  = (const float*)d_in[2];
  const float* Wo  = (const float*)d_in[3];
  const float* bo  = (const float*)d_in[4];
  const float* g1  = (const float*)d_in[5];
  const float* b1  = (const float*)d_in[6];
  const float* Wf1 = (const float*)d_in[7];
  const float* bf1 = (const float*)d_in[8];
  const float* Wf2 = (const float*)d_in[9];
  const float* bf2 = (const float*)d_in[10];
  const float* g2  = (const float*)d_in[11];
  const float* b2  = (const float*)d_in[12];
  float* out = (float*)d_out;

  // Workspace layout (aliased; ~106 MiB total)
  const size_t MB = 1ull << 20;
  char* ws = (char*)d_ws;
  __bf16* Xbf   = (__bf16*)(ws + 0);         // 16 MiB (reused as ff1bf later)
  __bf16* WkT   = (__bf16*)(ws + 16 * MB);   //  2 MiB
  __bf16* WvT   = (__bf16*)(ws + 18 * MB);   //  2 MiB
  __bf16* WoT   = (__bf16*)(ws + 20 * MB);   //  2 MiB
  __bf16* Wf1T  = (__bf16*)(ws + 22 * MB);   //  2 MiB
  __bf16* Wf2T  = (__bf16*)(ws + 24 * MB);   //  2 MiB
  __bf16* Kbf   = (__bf16*)(ws + 26 * MB);   // 16 MiB (reused as normBF later)
  __bf16* VTbf  = (__bf16*)(ws + 42 * MB);   // 16 MiB \ reused together as
  __bf16* attnbf= (__bf16*)(ws + 58 * MB);   // 16 MiB / normF (32 MiB f32)
  float*  y1    = (float*) (ws + 74 * MB);   // 32 MiB (reused as y2)
  __bf16* normBF= Kbf;
  float*  normF = (float*)(ws + 42 * MB);
  __bf16* ff1bf = Xbf;
  float*  y2    = y1;

  const int ND = kM * kD;  // 8192 * 1024

  // 1) Precision conversion + weight transposes (to [N,K] bf16)
  convert_bf16_k<<<(ND + 255) / 256, 256, 0, stream>>>(x, Xbf, ND);
  {
    dim3 g((kD * kHS + 255) / 256, kH);
    transpose_bf16_k<<<g, 256, 0, stream>>>(Wk, WkT, kD, kHS);
    transpose_bf16_k<<<g, 256, 0, stream>>>(Wv, WvT, kD, kHS);
  }
  {
    dim3 g((kD * kD + 255) / 256, 1);
    transpose_bf16_k<<<g, 256, 0, stream>>>(Wo,  WoT,  kD, kD);
    transpose_bf16_k<<<g, 256, 0, stream>>>(Wf1, Wf1T, kD, kD);
    transpose_bf16_k<<<g, 256, 0, stream>>>(Wf2, Wf2T, kD, kD);
  }

  // 2) K/V projections (q == k)
  proj_kv_k<<<dim3(kM / 128, kH), 128, 0, stream>>>(Xbf, WkT, WvT, Kbf, VTbf);

  // 3) Causal attention -> concatenated heads (bf16)
  attention_k<<<dim3(kT / 128, kB * kH), 256, 0, stream>>>(Kbf, VTbf, attnbf);

  // 4) Output projection + bias + residual x -> y1 (f32)
  gemm_bf16_k<<<dim3(kM / 128, kD / 256), 256, 0, stream>>>(
      attnbf, WoT, bo, x, nullptr, y1, nullptr, 0);

  // 5) LN1 -> norm (f32 for residual, bf16 for next GEMM)
  layernorm_k<<<kM, 256, 0, stream>>>(y1, g1, b1, normF, normBF);

  // 6) FFN layer 1: relu(norm @ Wf1 + bf1) -> bf16
  gemm_bf16_k<<<dim3(kM / 128, kD / 256), 256, 0, stream>>>(
      normBF, Wf1T, bf1, nullptr, nullptr, nullptr, ff1bf, 1);

  // 7) FFN layer 2 + bias + residuals (norm + x) -> y2 (f32)
  gemm_bf16_k<<<dim3(kM / 128, kD / 256), 256, 0, stream>>>(
      ff1bf, Wf2T, bf2, normF, x, y2, nullptr, 0);

  // 8) LN2 -> final output (f32)
  layernorm_k<<<kM, 256, 0, stream>>>(y2, g2, b2, out, nullptr);
}
